// LTCModel_85564338471698
// MI455X (gfx1250) — matrix operand order repaired and checked by reading, not compiled, not run on hardware
//
#include <hip/hip_runtime.h>
#include <hip/hip_bf16.h>

// ---- problem constants (from reference) ------------------------------------
#define B_    64
#define T_    2048
#define DIN   64
#define U_    128
#define NUNF  6
#define HSTR  136   // padded LDS row stride in f16 elems (272B = 68 dwords -> conflict-free)

typedef __attribute__((ext_vector_type(16))) _Float16 v16h;
typedef __attribute__((ext_vector_type(8)))  _Float16 v8h;
typedef __attribute__((ext_vector_type(8)))  float    v8f;
typedef __attribute__((ext_vector_type(4)))  float    f4;

union AVec { v16h v; v8h h[2]; };

// A-operand (16x32 f16) per ISA layout: lanes 0-15 = rows M, regs 0-3 hold
// K=kb+8*hi..+7 (pairs), regs 4-7 hold K=kb+16+8*hi..+7.
__device__ __forceinline__ v16h loadA_lds(const _Float16* hrow, int kb, int hi) {
    AVec a;
    a.h[0] = *(const v8h*)(hrow + kb + 8 * hi);
    a.h[1] = *(const v8h*)(hrow + kb + 16 + 8 * hi);
    return a.v;
}

// pack 4 float4 quads (16 consecutive-K f32 values split in two 8-wide runs)
// into a v16h A-operand chunk
__device__ __forceinline__ v16h packA(const f4* q) {
    v16h a;
#pragma unroll
    for (int i = 0; i < 4; ++i) {
        a[i]      = (_Float16)q[0][i];
        a[4 + i]  = (_Float16)q[1][i];
        a[8 + i]  = (_Float16)q[2][i];
        a[12 + i] = (_Float16)q[3][i];
    }
    return a;
}

// fast sigmoid: v_exp_f32 (via __expf) + v_rcp_f32 — no IEEE div expansion
__device__ __forceinline__ float fast_sigmoid(float z) {
    return __builtin_amdgcn_rcpf(1.0f + __expf(-z));
}

__global__ __launch_bounds__(256, 1)
void ltc_scan_fused(const float* __restrict__ x,
                    const float* __restrict__ in_kernel,
                    const float* __restrict__ rec_kernel,
                    const float* __restrict__ bias,
                    const float* __restrict__ tau,
                    const float* __restrict__ Arev,
                    const float* __restrict__ out_kernel,
                    const float* __restrict__ out_bias,
                    float* __restrict__ out)
{
    __shared__ _Float16 hbuf[2][16 * HSTR];   // double-buffered h state (f16 operand copy)

    const int tid  = threadIdx.x;
    const int lane = tid & 31;
    const int w    = tid >> 5;      // wave id 0..7: owns units n0 = 16w .. 16w+15
    const int nl   = lane & 15;
    const int hi   = lane >> 4;     // half-wave select (ISA lane-group)
    const int g    = blockIdx.x;    // batch stripe: rows 16g .. 16g+15
    const int n    = w * 16 + nl;   // unit owned by this lane in C/D (accumulator) layout
    const int m    = nl;            // A-operand row for this lane

    // ---- per-lane scalar constants ----------------------------------------
    const float dtc  = 1.0f / (float)NUNF;
    const float dtAn = dtc * Arev[n];            // dt * A[n]
    const float den0 = 1.0f + dtc / tau[n];      // 1 + dt/tau[n]  (one-time full div)
    const float bn   = bias[n];
    const float obv  = out_bias[0];
    float okr[8];
#pragma unroll
    for (int j = 0; j < 8; ++j) okr[j] = out_kernel[nl + 16 * j];

    // ---- one-time: weight B-operand tiles, f32 -> f16, resident in VGPRs --
    // B-operand (32x16 f16) layout: lane col = n, regs 0-3 hold K=kb+8*hi..+7,
    // regs 4-7 hold K=kb+16+8*hi..+7 (matches A-operand order).
    v16h Brec[4];
#pragma unroll
    for (int c = 0; c < 4; ++c) {
        const int kb = 32 * c + 8 * hi;
#pragma unroll
        for (int r = 0; r < 8; ++r) {
            const int k = kb + 2 * r + (r >= 4 ? 8 : 0);
            Brec[c][2 * r]     = (_Float16)rec_kernel[(size_t)k * U_ + n];
            Brec[c][2 * r + 1] = (_Float16)rec_kernel[(size_t)(k + 1) * U_ + n];
        }
    }
    v16h Bin[2];
#pragma unroll
    for (int c = 0; c < 2; ++c) {
        const int kb = 32 * c + 8 * hi;
#pragma unroll
        for (int r = 0; r < 8; ++r) {
            const int k = kb + 2 * r + (r >= 4 ? 8 : 0);
            Bin[c][2 * r]     = (_Float16)in_kernel[(size_t)k * U_ + n];
            Bin[c][2 * r + 1] = (_Float16)in_kernel[(size_t)(k + 1) * U_ + n];
        }
    }

    // ---- init state: h = 0 (f32 regs) + seed LDS buffer 0 -----------------
    float h[8];
#pragma unroll
    for (int r = 0; r < 8; ++r) {
        h[r] = 0.0f;
        hbuf[0][(r + 8 * hi) * HSTR + n] = (_Float16)0.0f;
    }
    __syncthreads();
    int cur = 0;

    // x row for this lane's batch element; 8 b128 quads per timestep
    const float* xbase = x + (size_t)(g * 16 + m) * T_ * DIN;

    f4 xq[8], xqn[8];
    {
        const f4* row = (const f4*)xbase;
        const int o = 2 * hi;
        xq[0] = row[o + 0];  xq[1] = row[o + 1];
        xq[2] = row[o + 4];  xq[3] = row[o + 5];
        xq[4] = row[o + 8];  xq[5] = row[o + 9];
        xq[6] = row[o + 12]; xq[7] = row[o + 13];
    }

    for (int t = 0; t < T_; ++t) {
        // prefetch x for t+1 (overlaps with the unfold chain below)
        {
            const int tn = (t + 1 < T_) ? (t + 1) : t;
            const f4* row = (const f4*)(xbase + (size_t)tn * DIN);
            const int o = 2 * hi;
            xqn[0] = row[o + 0];  xqn[1] = row[o + 1];
            xqn[2] = row[o + 4];  xqn[3] = row[o + 5];
            xqn[4] = row[o + 8];  xqn[5] = row[o + 9];
            xqn[6] = row[o + 12]; xqn[7] = row[o + 13];
        }

        // ---- xp = x_t @ kernel + bias: seed WMMA accumulator with bias ----
        v8f p;
#pragma unroll
        for (int r = 0; r < 8; ++r) p[r] = bn;
        {
            v16h a0 = packA(&xq[0]);
            p = __builtin_amdgcn_wmma_f32_16x16x32_f16(false, a0, false, Bin[0],
                                                       (short)0, p, false, false);
            v16h a1 = packA(&xq[4]);
            p = __builtin_amdgcn_wmma_f32_16x16x32_f16(false, a1, false, Bin[1],
                                                       (short)0, p, false, false);
        }

        // ---- 6 fused semi-implicit Euler unfolds --------------------------
#pragma unroll
        for (int u = 0; u < NUNF; ++u) {
            const _Float16* hrow = &hbuf[cur][m * HSTR];
            v8f s = p;   // seed accumulator with xp: WMMA yields z = xp + h@W
#pragma unroll
            for (int c = 0; c < 4; ++c) {
                v16h a = loadA_lds(hrow, 32 * c, hi);
                s = __builtin_amdgcn_wmma_f32_16x16x32_f16(false, a, false, Brec[c],
                                                           (short)0, s, false, false);
            }
            _Float16* dst = &hbuf[cur ^ 1][0];
#pragma unroll
            for (int r = 0; r < 8; ++r) {
                const float f   = fast_sigmoid(s[r]);                // v_exp + v_rcp
                const float num = fmaf(f, dtAn, h[r]);               // h + dt*f*A
                const float den = fmaf(dtc, f, den0);                // 1 + dt*(1/tau + f)
                const float hn  = num * __builtin_amdgcn_rcpf(den);
                h[r] = hn;                                           // f32 state stays in regs
                dst[(r + 8 * hi) * HSTR + n] = (_Float16)hn;         // operand copy -> LDS
            }
            __syncthreads();
            cur ^= 1;
        }

        // ---- fused output projection: out[b,t] = h . ok + ob --------------
        // wave w reduces rows 2w (lanes 0-15) and 2w+1 (lanes 16-31)
        {
            const int row = 2 * w + hi;
            const _Float16* hr = &hbuf[cur][row * HSTR];
            float acc = 0.0f;
#pragma unroll
            for (int j = 0; j < 8; ++j)
                acc += (float)hr[nl + 16 * j] * okr[j];
            acc += __shfl_xor(acc, 1, 32);
            acc += __shfl_xor(acc, 2, 32);
            acc += __shfl_xor(acc, 4, 32);
            acc += __shfl_xor(acc, 8, 32);
            if (nl == 0)
                out[(size_t)(g * 16 + row) * T_ + t] = acc + obv;
        }

#pragma unroll
        for (int i = 0; i < 8; ++i) xq[i] = xqn[i];
    }
}

extern "C" void kernel_launch(void* const* d_in, const int* in_sizes, int n_in,
                              void* d_out, int out_size, void* d_ws, size_t ws_size,
                              hipStream_t stream) {
    (void)in_sizes; (void)n_in; (void)d_ws; (void)ws_size; (void)out_size;
    const float* x   = (const float*)d_in[0];
    const float* ker = (const float*)d_in[1];
    const float* rk  = (const float*)d_in[2];
    const float* bs  = (const float*)d_in[3];
    const float* tau = (const float*)d_in[4];
    const float* A   = (const float*)d_in[5];
    const float* ok  = (const float*)d_in[6];
    const float* ob  = (const float*)d_in[7];
    float* out = (float*)d_out;

    ltc_scan_fused<<<dim3(B_ / 16), dim3(256), 0, stream>>>(
        x, ker, rk, bs, tau, A, ok, ob, out);
}
